// _GIN_77086073028963
// MI455X (gfx1250) — compile-verified, matrix operand order checked by reading, not run on hardware
//
#include <hip/hip_runtime.h>
#include <hip/hip_bf16.h>
#include <stddef.h>

// ---------------------------------------------------------------------------
// GIN forward for MI455X (gfx1250, wave32, WMMA).
// GEMMs use v_wmma_f32_16x16x32_bf16 (fp32 accumulate), weights staged
// transposed in LDS as bf16 so B-fragments are contiguous ds_load_b128.
// Scatter-add / pooling use native f32 global atomics (unsafeAtomicAdd ->
// global_atomic_add_f32; the 192MB L2's atomic units do the RMW in-cache).
// Workspace layout (floats): A[50000*128] | Z[50000*128] | T[50000*128] |
//                            sums[64*128] | counts[64]   (~77 MB)
// ---------------------------------------------------------------------------

typedef __attribute__((ext_vector_type(16))) __bf16 v16bf;
typedef __attribute__((ext_vector_type(8)))  float  v8f;

__device__ __forceinline__ v8f wmma_bf16(v16bf a, v16bf b, v8f c) {
  // (neg_a, A, neg_b, B, c_mod, C, reuse_a, reuse_b)
  return __builtin_amdgcn_wmma_f32_16x16x32_bf16(
      false, a, false, b, (short)0, c, false, false);
}

// Native f32 global atomic add (global_atomic_add_f32, no CAS fallback).
__device__ __forceinline__ void atomAddF32(float* p, float v) {
#if defined(__HIP_DEVICE_COMPILE__)
  unsafeAtomicAdd(p, v);
#else
  atomicAdd(p, v);
#endif
}

#define GIN_N   50000
#define GIN_E   640000
#define GIN_D   128
#define GIN_G   64
#define GIN_C   32
#define GIN_L   4

// ---------------------------------------------------------------- copy z = h
__global__ void k_copy(const float4* __restrict__ src, float4* __restrict__ dst,
                       int n4) {
  int i = blockIdx.x * blockDim.x + threadIdx.x;
  if (i < n4) dst[i] = src[i];
}

// ---------------------------------------------------------------- zero
__global__ void k_zero(float* __restrict__ p, int n) {
  int i = blockIdx.x * blockDim.x + threadIdx.x;
  if (i < n) p[i] = 0.0f;
}

// ----------------------------------------------- scatter-add: z[dst] += h[src]
// one (edge, 4-dim quad) per lane: coalesced float4 gather, 4 float atomics.
__global__ void k_scatter(const float* __restrict__ h,
                          const int* __restrict__ ei,   // [2, E] int32
                          float* __restrict__ z, int nE) {
  int tid = blockIdx.x * blockDim.x + threadIdx.x;
  int e = tid >> 5;
  if (e >= nE) return;
  if (e + 4096 < nE) __builtin_prefetch(ei + e + 4096, 0, 1);  // global_prefetch
  int q = (tid & 31) * 4;
  int s = ei[e];
  int d = ei[nE + e];
  const float4 v = *(const float4*)(h + (size_t)s * GIN_D + q);
  float* zp = z + (size_t)d * GIN_D + q;
  atomAddF32(zp + 0, v.x);
  atomAddF32(zp + 1, v.y);
  atomAddF32(zp + 2, v.z);
  atomAddF32(zp + 3, v.w);
}

// ------------------------------------------------------------------- WMMA GEMM
// Out[M x 128] = act( A[M x 128] @ W[128 x 128] + bias ), bf16 inputs, f32 acc.
// 8 waves / WG; wave owns one 16-row tile, computes all 8 column tiles.
__global__ __launch_bounds__(256) void k_gemm(const float* __restrict__ A,
                                              const float* __restrict__ W,
                                              const float* __restrict__ bias,
                                              float* __restrict__ Out,
                                              int M, int relu) {
  __shared__ __bf16 lWT[GIN_D][GIN_D + 8];  // W transposed: lWT[n][k], padded

  const int t = threadIdx.x;
  // stage W (row-major K x N) transposed into LDS as bf16, coalesced reads
  #pragma unroll 4
  for (int i = 0; i < 64; ++i) {
    int idx = t + i * 256;          // 0 .. 16383
    int k = idx >> 7, n = idx & 127;
    lWT[n][k] = (__bf16)W[idx];
  }
  __syncthreads();

  const int wave = t >> 5, lane = t & 31;
  const int mTile = blockIdx.x * 8 + wave;
  if (mTile * 16 >= M) return;

  const int rowSel = lane & 15;          // row within M-tile / col within N-tile
  const int koff   = (lane >> 4) * 8;    // 16-bit A/B K-layout half selector
  const size_t row = (size_t)(mTile * 16 + rowSel);

  v8f acc[8];
  #pragma unroll
  for (int n = 0; n < 8; ++n) acc[n] = (v8f){};

  #pragma unroll
  for (int ks = 0; ks < 4; ++ks) {
    const int k0 = ks * 32;
    // A fragment (16x32 bf16): lane<16 -> K {k0..k0+7, k0+16..k0+23}
    //                          lane>=16 -> +8 on both halves (koff)
    union { v16bf v; __bf16 e[16]; } af;
    const float* ap = A + row * GIN_D + k0 + koff;
    #pragma unroll
    for (int i = 0; i < 8; ++i) af.e[i]     = (__bf16)ap[i];
    #pragma unroll
    for (int i = 0; i < 8; ++i) af.e[8 + i] = (__bf16)ap[16 + i];

    #pragma unroll
    for (int nt = 0; nt < 8; ++nt) {
      union { v16bf v; __bf16 e[16]; } bfr;
      const __bf16* bp = &lWT[nt * 16 + rowSel][k0 + koff];
      #pragma unroll
      for (int i = 0; i < 8; ++i) bfr.e[i]     = bp[i];
      #pragma unroll
      for (int i = 0; i < 8; ++i) bfr.e[8 + i] = bp[16 + i];
      acc[nt] = wmma_bf16(af.v, bfr.v, acc[nt]);
    }
  }

  // store: C layout -> element (m = r + 8*(lane>>4), n = lane&15) in VGPR r
  const int mBase = mTile * 16 + (lane >> 4) * 8;
  #pragma unroll
  for (int nt = 0; nt < 8; ++nt) {
    const int n = nt * 16 + rowSel;
    const float b = bias[n];
    #pragma unroll
    for (int r = 0; r < 8; ++r) {
      float v = acc[nt][r] + b;
      if (relu) v = fmaxf(v, 0.0f);
      Out[(size_t)(mBase + r) * GIN_D + n] = v;
    }
  }
}

// ------------------------------------------------------- pooled sums / counts
__global__ void k_poolsum(const float* __restrict__ h,
                          const int* __restrict__ batch,
                          float* __restrict__ sums, float* __restrict__ counts,
                          int nNodes) {
  int tid = blockIdx.x * blockDim.x + threadIdx.x;
  int node = tid >> 5;
  if (node >= nNodes) return;
  int q = (tid & 31) * 4;
  int g = batch[node];
  const float4 v = *(const float4*)(h + (size_t)node * GIN_D + q);
  float* sp = sums + (size_t)g * GIN_D + q;
  atomAddF32(sp + 0, v.x);
  atomAddF32(sp + 1, v.y);
  atomAddF32(sp + 2, v.z);
  atomAddF32(sp + 3, v.w);
  if (q == 0) atomAddF32(&counts[g], 1.0f);
}

// ----------------------------- pooled = sums/counts ; logits = pooled@Wl + bl
// single WG: 8 waves = 4 M-tiles x 2 N-tiles of the 64x128x32 GEMM.
__global__ __launch_bounds__(256) void k_final(const float* __restrict__ sums,
                                               const float* __restrict__ counts,
                                               const float* __restrict__ wlin,
                                               const float* __restrict__ blin,
                                               float* __restrict__ pooled_out,
                                               float* __restrict__ logits_out) {
  __shared__ float  lP[GIN_G * GIN_D];       // pooled, f32 (32 KB)
  __shared__ __bf16 lWT[GIN_C][GIN_D + 8];   // w_lin transposed [n][k]

  const int t = threadIdx.x;
  #pragma unroll 4
  for (int i = 0; i < 32; ++i) {
    int idx = t + i * 256;                   // 0 .. 8191
    int g = idx >> 7;
    float c = fmaxf(counts[g], 1.0f);
    float v = sums[idx] / c;
    lP[idx] = v;
    pooled_out[idx] = v;
  }
  #pragma unroll 4
  for (int i = 0; i < 16; ++i) {
    int idx = t + i * 256;                   // 0 .. 4095, wlin is 128x32
    int k = idx >> 5, n = idx & 31;
    lWT[n][k] = (__bf16)wlin[idx];
  }
  __syncthreads();

  const int wave = t >> 5, lane = t & 31;
  const int mT = wave >> 1, nT = wave & 1;
  const int rowSel = lane & 15;
  const int koff = (lane >> 4) * 8;

  v8f acc = (v8f){};
  #pragma unroll
  for (int ks = 0; ks < 4; ++ks) {
    const int k0 = ks * 32;
    union { v16bf v; __bf16 e[16]; } af, bfr;
    const float* ap = lP + (size_t)(mT * 16 + rowSel) * GIN_D + k0 + koff;
    #pragma unroll
    for (int i = 0; i < 8; ++i) af.e[i]     = (__bf16)ap[i];
    #pragma unroll
    for (int i = 0; i < 8; ++i) af.e[8 + i] = (__bf16)ap[16 + i];
    const __bf16* bp = &lWT[nT * 16 + rowSel][k0 + koff];
    #pragma unroll
    for (int i = 0; i < 8; ++i) bfr.e[i]     = bp[i];
    #pragma unroll
    for (int i = 0; i < 8; ++i) bfr.e[8 + i] = bp[16 + i];
    acc = wmma_bf16(af.v, bfr.v, acc);
  }

  const int n = nT * 16 + rowSel;
  const float b = blin[n];
  const int m0 = mT * 16 + (lane >> 4) * 8;
  #pragma unroll
  for (int r = 0; r < 8; ++r)
    logits_out[(size_t)(m0 + r) * GIN_C + n] = acc[r] + b;
}

// ---------------------------------------------------------------------------
extern "C" void kernel_launch(void* const* d_in, const int* in_sizes, int n_in,
                              void* d_out, int out_size, void* d_ws,
                              size_t ws_size, hipStream_t stream) {
  const float* x     = (const float*)d_in[0];
  const int*   ei    = (const int*)d_in[1];   // edge_index [2, E]
  const int*   batch = (const int*)d_in[2];
  const float* w1    = (const float*)d_in[3]; // [L,128,128]
  const float* b1    = (const float*)d_in[4]; // [L,128]
  const float* w2    = (const float*)d_in[5];
  const float* b2    = (const float*)d_in[6];
  const float* wlin  = (const float*)d_in[7]; // [128,32]
  const float* blin  = (const float*)d_in[8]; // [32]
  float* out = (float*)d_out;

  const size_t nd = (size_t)GIN_N * GIN_D;
  float* bufA   = (float*)d_ws;       // h for layers >= 1 / GEMM2 output
  float* bufZ   = bufA + nd;          // z = h + agg
  float* bufT   = bufZ + nd;          // t = relu(z @ w1 + b1)
  float* sums   = bufT + nd;          // 64*128
  float* counts = sums + (size_t)GIN_G * GIN_D;  // 64

  const int n4 = (int)(nd / 4);
  const int copyBlocks = (n4 + 255) / 256;
  const int scatThreads = GIN_E * 32;
  const int scatBlocks = (scatThreads + 255) / 256;
  const int gemmBlocks = (GIN_N / 16 + 7) / 8;   // 3125 wave tiles -> 391 WGs

  const float* h = x;
  for (int l = 0; l < GIN_L; ++l) {
    k_copy<<<copyBlocks, 256, 0, stream>>>((const float4*)h, (float4*)bufZ, n4);
    k_scatter<<<scatBlocks, 256, 0, stream>>>(h, ei, bufZ, GIN_E);
    k_gemm<<<gemmBlocks, 256, 0, stream>>>(
        bufZ, w1 + (size_t)l * GIN_D * GIN_D, b1 + (size_t)l * GIN_D,
        bufT, GIN_N, 1);
    k_gemm<<<gemmBlocks, 256, 0, stream>>>(
        bufT, w2 + (size_t)l * GIN_D * GIN_D, b2 + (size_t)l * GIN_D,
        bufA, GIN_N, (l < GIN_L - 1) ? 1 : 0);
    h = bufA;
  }

  const int zn = GIN_G * GIN_D + GIN_G;
  k_zero<<<(zn + 255) / 256, 256, 0, stream>>>(sums, zn);
  k_poolsum<<<(GIN_N * 32 + 255) / 256, 256, 0, stream>>>(h, batch, sums,
                                                          counts, GIN_N);
  k_final<<<1, 256, 0, stream>>>(sums, counts, wlin, blin, out,
                                 out + (size_t)GIN_G * GIN_D);
}
